// PointNetSetAbstractionMsg_44074954391638
// MI455X (gfx1250) — compile-verified, matrix operand order checked by reading, not run on hardware
//
#include <hip/hip_runtime.h>
#include <hip/hip_bf16.h>

// ---------------------------------------------------------------------------
// PointNet++ Set Abstraction (MSG) for MI455X (gfx1250, wave32, WMMA bf16)
// ---------------------------------------------------------------------------
//   1. fps_kernel            : farthest-point sampling, 1 block / batch
//   2. gather_centroids      : new_xyz -> d_out[0 .. B*3*S)
//   3. per scale (r,K,mlp):
//        group_kernel        : ball query + feature grouping -> bf16 [M,32]
//        per layer:
//          pack_w_kernel     : f32 weights -> bf16, K-padded
//          zero_kernel       : clear BN stats accumulators
//          gemm_kernel<KP,NT>: X[M,KP] x W^T[N,KP] + bias -> Y bf16 (WMMA,
//                              W staged in LDS, full 16xN strip per wave)
//          bn_stats_kernel   : per-channel sum / sumsq
//          bn_fin_kernel     : fold (mu,var,gamma,beta,eps) -> scale/shift
//          bn_relu_kernel    : (not last) normalize+relu -> next X (bf16)
//          bn_relu_max_kernel: (last) normalize+relu+max over K -> d_out
// ---------------------------------------------------------------------------

typedef __attribute__((ext_vector_type(16))) __bf16 v16bf;
typedef __attribute__((ext_vector_type(8)))  float  v8f;

#define B_   8
#define N_   8192
#define S_   512
#define OUTCH_ 320

__device__ __forceinline__ unsigned short f2bf(float f) {
  unsigned u = __float_as_uint(f);
  u += 0x7FFFu + ((u >> 16) & 1u);          // round-to-nearest-even
  return (unsigned short)(u >> 16);
}
__device__ __forceinline__ float bf2f(unsigned short h) {
  return __uint_as_float(((unsigned)h) << 16);
}

// ---------------------------------------------------------------------------
// 1) Farthest point sampling: one block (1024 thr = 32 waves) per batch.
// ---------------------------------------------------------------------------
__global__ __launch_bounds__(1024)
void fps_kernel(const float* __restrict__ xyz, int* __restrict__ cent) {
  const int b = blockIdx.x;
  const float* xb = xyz + (size_t)b * 3 * N_;
  const int t = threadIdx.x;
  const int lane = t & 31, wave = t >> 5;

  float dist[8];
#pragma unroll
  for (int i = 0; i < 8; ++i) dist[i] = 1e10f;

  __shared__ float swv[32];
  __shared__ int   swi[32];
  __shared__ int   sfar;

  int far = 0;
  for (int it = 0; it < S_; ++it) {
    if (t == 0) cent[b * S_ + it] = far;
    const float cx = xb[far], cy = xb[N_ + far], cz = xb[2 * N_ + far];

    float bv = -1.0f; int bi = 0;
#pragma unroll
    for (int i = 0; i < 8; ++i) {
      const int j = t + i * 1024;
      const float dx = xb[j] - cx, dy = xb[N_ + j] - cy, dz = xb[2 * N_ + j] - cz;
      const float d = dx * dx + dy * dy + dz * dz;
      const float nd = fminf(dist[i], d);
      dist[i] = nd;
      if (nd > bv) { bv = nd; bi = j; }     // ascending j -> keeps first max
    }
#pragma unroll
    for (int off = 16; off > 0; off >>= 1) {
      const float ov = __shfl_down(bv, off);
      const int   oi = __shfl_down(bi, off);
      if (ov > bv || (ov == bv && oi < bi)) { bv = ov; bi = oi; }
    }
    if (lane == 0) { swv[wave] = bv; swi[wave] = bi; }
    __syncthreads();
    if (wave == 0) {
      float v = swv[lane]; int ix = swi[lane];
#pragma unroll
      for (int off = 16; off > 0; off >>= 1) {
        const float ov = __shfl_down(v, off);
        const int   oi = __shfl_down(ix, off);
        if (ov > v || (ov == v && oi < ix)) { v = ov; ix = oi; }
      }
      if (lane == 0) sfar = ix;
    }
    __syncthreads();
    far = sfar;
  }
}

// ---------------------------------------------------------------------------
// 2) Gather centroid coordinates into d_out (new_xyz, layout [B,3,S])
// ---------------------------------------------------------------------------
__global__ void gather_centroids(const float* __restrict__ xyz,
                                 const int* __restrict__ cent,
                                 float* __restrict__ newxyz) {
  const int t = blockIdx.x * blockDim.x + threadIdx.x;
  if (t >= B_ * S_) return;
  const int b = t >> 9, s = t & 511;
  const int j = cent[t];
#pragma unroll
  for (int d = 0; d < 3; ++d)
    newxyz[b * (3 * S_) + d * S_ + s] = xyz[((size_t)b * 3 + d) * N_ + j];
}

// ---------------------------------------------------------------------------
// 3) Ball query + grouping. One wave per centroid; ordered stream compaction.
// ---------------------------------------------------------------------------
__device__ __forceinline__ void write_feat_row(unsigned short* dst,
                                               float p0, float p1, float p2,
                                               float g0, float g1, float g2) {
  dst[0] = f2bf(p0); dst[1] = f2bf(p1); dst[2] = f2bf(p2);
  dst[3] = f2bf(g0); dst[4] = f2bf(g1); dst[5] = f2bf(g2);
#pragma unroll
  for (int c = 6; c < 32; ++c) dst[c] = 0;
}

__global__ void group_kernel(const float* __restrict__ xyz,
                             const float* __restrict__ pts,
                             const float* __restrict__ newxyz,
                             unsigned short* __restrict__ Xbuf,
                             int K, float r2) {
  const int wid  = (blockIdx.x * blockDim.x + threadIdx.x) >> 5;
  const int lane = threadIdx.x & 31;
  if (wid >= B_ * S_) return;
  const int b = wid / S_, s = wid % S_;

  const float cx = newxyz[b * (3 * S_) + 0 * S_ + s];
  const float cy = newxyz[b * (3 * S_) + 1 * S_ + s];
  const float cz = newxyz[b * (3 * S_) + 2 * S_ + s];
  const float* xb = xyz + (size_t)b * 3 * N_;
  const float* pb = pts + (size_t)b * 3 * N_;
  unsigned short* rowbase = Xbuf + (size_t)wid * K * 32;

  int cnt = 0, firstJ = -1;
  for (int j0 = 0; j0 < N_ && cnt < K; j0 += 32) {
    const int j = j0 + lane;
    const float dx = xb[j] - cx, dy = xb[N_ + j] - cy, dz = xb[2 * N_ + j] - cz;
    const float sq = dx * dx + dy * dy + dz * dz;
    const bool in = (sq <= r2);
    const unsigned mask = (unsigned)__ballot(in);
    if (firstJ < 0 && mask) firstJ = j0 + __ffs(mask) - 1;
    const int slot = cnt + __popc(mask & ((1u << lane) - 1u));
    if (in && slot < K)
      write_feat_row(rowbase + (size_t)slot * 32,
                     pb[j], pb[N_ + j], pb[2 * N_ + j], dx, dy, dz);
    cnt += __popc(mask);
  }
  if (cnt < K) {                       // pad with first in-range point
    const int j = (firstJ < 0) ? 0 : firstJ;
    const float dx = xb[j] - cx, dy = xb[N_ + j] - cy, dz = xb[2 * N_ + j] - cz;
    const float p0 = pb[j], p1 = pb[N_ + j], p2 = pb[2 * N_ + j];
    for (int p = cnt + lane; p < K; p += 32)
      write_feat_row(rowbase + (size_t)p * 32, p0, p1, p2, dx, dy, dz);
  }
}

// ---------------------------------------------------------------------------
// Weight pack: f32 [C, Cin] -> bf16 [C, Kp] with zero padding (Kp % 32 == 0)
// ---------------------------------------------------------------------------
__global__ void pack_w_kernel(const float* __restrict__ W,
                              unsigned short* __restrict__ Wbf,
                              int C, int Cin, int Kp) {
  const int t = blockIdx.x * blockDim.x + threadIdx.x;
  if (t >= C * Kp) return;
  const int o = t / Kp, k = t % Kp;
  Wbf[t] = (k < Cin) ? f2bf(W[o * Cin + k]) : (unsigned short)0;
}

__global__ void zero_kernel(float* __restrict__ p, int n) {
  const int t = blockIdx.x * blockDim.x + threadIdx.x;
  if (t < n) p[t] = 0.0f;
}

// ---------------------------------------------------------------------------
// 4) GEMM + bias: Y[M,N] = X[M,KP] * Wbf[N,KP]^T + bias   (N = NT*16)
//    - Whole W tile staged in LDS once per 256-thread block (max 32 KB).
//    - Each wave computes a full 16 x N row-strip: A fragment loaded once
//      per k-step from global, reused across all NT B-tiles (ds_load).
//    CDNA5 16-bit fragment layout: lane half h holds K = {h*8.., 16+h*8..}
//    as two contiguous 16-byte loads.
// ---------------------------------------------------------------------------
template <int KP, int NT>
__global__ __launch_bounds__(256)
void gemm_kernel(const unsigned short* __restrict__ X,
                 const unsigned short* __restrict__ Wbf,
                 const float* __restrict__ bias,
                 unsigned short* __restrict__ Y,
                 int M) {
  constexpr int NDIM = NT * 16;
  __shared__ unsigned short smemW[NDIM * KP];

  // cooperative W load (uint4 = 8 bf16)
  {
    const int nvec = (NDIM * KP) / 8;
    for (int i = threadIdx.x; i < nvec; i += 256)
      ((uint4*)smemW)[i] = ((const uint4*)Wbf)[i];
  }
  __syncthreads();

  const int wave = threadIdx.x >> 5;
  const int lane = threadIdx.x & 31;
  const int tm   = blockIdx.x * 8 + wave;        // M/16 always divisible by 8
  const int half = lane >> 4, l16 = lane & 15;

  const unsigned short* xrow = X + (size_t)(tm * 16 + l16) * KP;

  v8f acc[NT];
#pragma unroll
  for (int t = 0; t < NT; ++t) acc[t] = (v8f){};

#pragma unroll
  for (int k = 0; k < KP; k += 32) {
    __builtin_prefetch(xrow + k + 64, 0, 0);     // global_prefetch_b8
    union { v16bf v; uint4 u[2]; } a;
    a.u[0] = *(const uint4*)(xrow + k + half * 8);
    a.u[1] = *(const uint4*)(xrow + k + 16 + half * 8);
#pragma unroll
    for (int tn = 0; tn < NT; ++tn) {
      union { v16bf v; uint4 u[2]; } bfrag;
      const unsigned short* wl = smemW + (tn * 16 + l16) * KP + k + half * 8;
      bfrag.u[0] = *(const uint4*)wl;            // ds_load_b128
      bfrag.u[1] = *(const uint4*)(wl + 16);
      acc[tn] = __builtin_amdgcn_wmma_f32_16x16x32_bf16(
          false, a.v, false, bfrag.v, (short)0, acc[tn], false, false);
    }
  }

#pragma unroll
  for (int tn = 0; tn < NT; ++tn) {
    const int col = tn * 16 + l16;
    const float bs = bias[col];
    unsigned short* ybase = Y + (size_t)(tm * 16 + half * 8) * NDIM + col;
#pragma unroll
    for (int i = 0; i < 8; ++i)                  // D: row = tm*16 + half*8 + i
      ybase[(size_t)i * NDIM] = f2bf(acc[tn][i] + bs);
  }
}

// ---------------------------------------------------------------------------
// 5) BN statistics: per-channel sum / sumsq. blockDim == N (32..128).
// ---------------------------------------------------------------------------
__global__ void bn_stats_kernel(const unsigned short* __restrict__ Y,
                                float* __restrict__ stats, int M, int N) {
  const int c = threadIdx.x;            // channel
  float s1 = 0.0f, s2 = 0.0f;
  for (int r = blockIdx.x; r < M; r += gridDim.x) {
    const float y = bf2f(Y[(size_t)r * N + c]);
    s1 += y;
    s2 += y * y;
  }
  atomicAdd(&stats[c], s1);
  atomicAdd(&stats[N + c], s2);
}

__global__ void bn_fin_kernel(const float* __restrict__ stats,
                              const float* __restrict__ gamma,
                              const float* __restrict__ beta,
                              float* __restrict__ musig, int N, float invM) {
  const int c = threadIdx.x;
  if (c >= N) return;
  const float mu  = stats[c] * invM;
  const float var = fmaxf(stats[N + c] * invM - mu * mu, 0.0f);
  const float rs  = rsqrtf(var + 1e-5f);
  const float sc  = gamma[c] * rs;
  musig[c]     = sc;                    // scale
  musig[N + c] = beta[c] - mu * sc;     // shift
}

// ---------------------------------------------------------------------------
// 6a) BN + ReLU -> next layer input (bf16)
// ---------------------------------------------------------------------------
__global__ void bn_relu_kernel(const unsigned short* __restrict__ Y,
                               const float* __restrict__ musig,
                               unsigned short* __restrict__ Xn,
                               size_t total, int N) {
  const size_t t = (size_t)blockIdx.x * blockDim.x + threadIdx.x;
  if (t >= total) return;
  const int c = (int)(t % N);
  const float v = fmaf(bf2f(Y[t]), musig[c], musig[N + c]);
  Xn[t] = f2bf(fmaxf(v, 0.0f));
}

// ---------------------------------------------------------------------------
// 6b) BN + ReLU + max over K -> d_out new_points [B, 320, S]
// ---------------------------------------------------------------------------
__global__ void bn_relu_max_kernel(const unsigned short* __restrict__ Y,
                                   const float* __restrict__ musig,
                                   float* __restrict__ newpts,
                                   int K, int N, int ch_off) {
  const int t = blockIdx.x * blockDim.x + threadIdx.x;
  if (t >= B_ * S_ * N) return;
  const int c = t % N;
  const int s = (t / N) % S_;
  const int b = t / (N * S_);
  const float sc = musig[c], sh = musig[N + c];
  const size_t base = ((size_t)(b * S_ + s) * K) * N + c;
  float m = 0.0f;                       // post-ReLU values are >= 0
  for (int k = 0; k < K; ++k) {
    const float v = fmaf(bf2f(Y[base + (size_t)k * N]), sc, sh);
    m = fmaxf(m, fmaxf(v, 0.0f));
  }
  newpts[(size_t)b * (OUTCH_ * S_) + (size_t)(ch_off + c) * S_ + s] = m;
}

// ---------------------------------------------------------------------------
// Host: orchestrate the whole SA-MSG forward on `stream`.
// ---------------------------------------------------------------------------
extern "C" void kernel_launch(void* const* d_in, const int* in_sizes, int n_in,
                              void* d_out, int out_size, void* d_ws, size_t ws_size,
                              hipStream_t stream) {
  const float* xyz = (const float*)d_in[0];
  const float* pts = (const float*)d_in[1];

  float* out    = (float*)d_out;
  float* newxyz = out;                         // [B,3,S]
  float* newpts = out + (size_t)B_ * 3 * S_;   // [B,320,S]

  char* ws = (char*)d_ws;
  int*            cent  = (int*)(ws);                              // 16 KB
  unsigned short* wbf   = (unsigned short*)(ws + 16384);           // 32 KB
  float*          stats = (float*)(ws + 49152);                    // 1 KB
  float*          musig = (float*)(ws + 50176);                    // 1 KB
  unsigned short* bufA  = (unsigned short*)(ws + 65536);           // 134 MB
  unsigned short* bufB  = (unsigned short*)(ws + 65536 + ((size_t)1 << 27));

  // 1) FPS + centroid gather
  fps_kernel<<<B_, 1024, 0, stream>>>(xyz, cent);
  gather_centroids<<<(B_ * S_ + 255) / 256, 256, 0, stream>>>(xyz, cent, newxyz);

  const int   KS[3]     = {16, 32, 128};
  const float R2[3]     = {0.25f, 1.0f, 4.0f};
  const int   MLP[3][3] = {{32, 32, 64}, {64, 64, 128}, {64, 96, 128}};

  int ch_off = 0;
  for (int sc = 0; sc < 3; ++sc) {
    const int K = KS[sc];
    const int M = B_ * S_ * K;

    // 2) ball query + grouping -> bufA [M, 32] bf16
    {
      const int waves = B_ * S_;
      group_kernel<<<(waves * 32 + 255) / 256, 256, 0, stream>>>(
          xyz, pts, newxyz, bufA, K, R2[sc]);
    }

    int cin = 6, kp = 32;
    for (int l = 0; l < 3; ++l) {
      const int cout = MLP[sc][l];
      const int pidx = 2 + sc * 12 + l * 4;
      const float* W     = (const float*)d_in[pidx + 0];
      const float* bias  = (const float*)d_in[pidx + 1];
      const float* gamma = (const float*)d_in[pidx + 2];
      const float* beta  = (const float*)d_in[pidx + 3];

      pack_w_kernel<<<(cout * kp + 255) / 256, 256, 0, stream>>>(W, wbf, cout, cin, kp);
      zero_kernel<<<1, 256, 0, stream>>>(stats, 2 * cout);

      // 3) WMMA GEMM + bias -> bufB [M, cout] bf16 (strip-mined, W in LDS)
      {
        const int blocks = (M / 16) / 8;       // 8 waves per block, 1 tm each
        if      (kp == 32 && cout ==  32) gemm_kernel<32, 2><<<blocks, 256, 0, stream>>>(bufA, wbf, bias, bufB, M);
        else if (kp == 32 && cout ==  64) gemm_kernel<32, 4><<<blocks, 256, 0, stream>>>(bufA, wbf, bias, bufB, M);
        else if (kp == 64 && cout ==  64) gemm_kernel<64, 4><<<blocks, 256, 0, stream>>>(bufA, wbf, bias, bufB, M);
        else if (kp == 64 && cout ==  96) gemm_kernel<64, 6><<<blocks, 256, 0, stream>>>(bufA, wbf, bias, bufB, M);
        else if (kp == 64 && cout == 128) gemm_kernel<64, 8><<<blocks, 256, 0, stream>>>(bufA, wbf, bias, bufB, M);
        else if (kp == 96 && cout == 128) gemm_kernel<96, 8><<<blocks, 256, 0, stream>>>(bufA, wbf, bias, bufB, M);
      }

      // 4) BN stats + finalize
      bn_stats_kernel<<<2048, cout, 0, stream>>>(bufB, stats, M, cout);
      bn_fin_kernel<<<1, 128, 0, stream>>>(stats, gamma, beta, musig, cout, 1.0f / (float)M);

      if (l < 2) {
        const size_t total = (size_t)M * cout;
        bn_relu_kernel<<<(unsigned)((total + 255) / 256), 256, 0, stream>>>(
            bufB, musig, bufA, total, cout);
      } else {
        bn_relu_max_kernel<<<(B_ * S_ * cout + 255) / 256, 256, 0, stream>>>(
            bufB, musig, newpts, K, cout, ch_off);
      }
      cin = cout;
      kp = cout;   // all MLP widths are multiples of 32
    }
    ch_off += MLP[sc][2];
  }
}
